// SCE_LRD_67053029425342
// MI455X (gfx1250) — matrix operand order repaired and checked by reading, not compiled, not run on hardware
//
#include <hip/hip_runtime.h>
#include <hip/hip_bf16.h>
#include <math.h>

typedef __attribute__((ext_vector_type(16))) __bf16          v16bf;
typedef __attribute__((ext_vector_type(16))) unsigned short  v16u;
typedef __attribute__((ext_vector_type(8)))  float           v8f;

#define B_     32
#define C_     256
#define HW_    4096
#define NPAIR  36   // 8*9/2 upper-triangular 32x32 block-pairs
#define NGRP   16   // batch groups (fast path)
#define BPG2   2    // batches per group (fast path)
#define KSPLIT 2    // K-range split (fast path)
#define NSLICE (NGRP * KSPLIT)   // 32 partial slices (fast path)
#define NPARTF 8    // groups for fallback path
#define BPGF   4

struct U16x16 { uint4 lo, hi; };   // 32 bytes == v16bf

// float -> bf16 round-to-nearest-even (bit math; only in mem-bound kernels)
__device__ __forceinline__ unsigned short f2bf(float x) {
    unsigned u = __float_as_uint(x);
    u += 0x7FFFu + ((u >> 16) & 1u);
    return (unsigned short)(u >> 16);
}

// ---------------- Kernel 1: per-(b,c) stats: a=sigmoid(mean), inv_norm, max ----
__global__ void stats_kernel(const float* __restrict__ F, float* __restrict__ a,
                             float* __restrict__ invn, float* __restrict__ mx) {
    int bc = blockIdx.x;
    int t  = threadIdx.x;
    const float4* p4 = (const float4*)(F + (size_t)bc * HW_);
    float s = 0.f, sq = 0.f, m = -INFINITY;
#pragma unroll
    for (int i = 0; i < 4; ++i) {
        float4 v = p4[t * 4 + i];
        s  += v.x + v.y + v.z + v.w;
        sq += v.x*v.x + v.y*v.y + v.z*v.z + v.w*v.w;
        m   = fmaxf(m, fmaxf(fmaxf(v.x, v.y), fmaxf(v.z, v.w)));
    }
    __shared__ float rs[256], rq[256], rm[256];
    rs[t] = s; rq[t] = sq; rm[t] = m;
    __syncthreads();
    for (int off = 128; off > 0; off >>= 1) {
        if (t < off) {
            rs[t] += rs[t + off];
            rq[t] += rq[t + off];
            rm[t]  = fmaxf(rm[t], rm[t + off]);
        }
        __syncthreads();
    }
    if (t == 0) {
        float mean = rs[0] * (1.0f / (float)HW_);
        a[bc]    = 1.0f / (1.0f + expf(-mean));
        invn[bc] = 1.0f / (sqrtf(rq[0]) + 1e-6f);
        mx[bc]   = rm[0];
    }
}

// ---------------- Kernel 2: 5x5 box-count density -> s = sqrt(a*d) ------------
__global__ void density_kernel(const float* __restrict__ F, const float* __restrict__ a,
                               const float* __restrict__ mx, float* __restrict__ s_out) {
    int bc = blockIdx.x;
    int t  = threadIdx.x;
    const float4* p4 = (const float4*)(F + (size_t)bc * HW_);
    float thr = 0.6f * mx[bc];
    __shared__ unsigned char bin[HW_];
#pragma unroll
    for (int i = 0; i < 4; ++i) {
        float4 v = p4[t * 4 + i];
        int base = t * 16 + i * 4;
        bin[base + 0] = (v.x >= thr);
        bin[base + 1] = (v.y >= thr);
        bin[base + 2] = (v.z >= thr);
        bin[base + 3] = (v.w >= thr);
    }
    __syncthreads();
    int hits = 0;
    for (int i = 0; i < 16; ++i) {
        int p = t * 16 + i;
        int y = p >> 6, x = p & 63;
        int cnt = 0;
        for (int dy = -2; dy <= 2; ++dy) {
            int yy = y + dy;
            if (yy < 0 || yy > 63) continue;
            for (int dx = -2; dx <= 2; ++dx) {
                int xx = x + dx;
                if (xx < 0 || xx > 63) continue;
                cnt += bin[yy * 64 + xx];
            }
        }
        hits += (cnt >= 13);    // density >= 0.5  <=>  count >= 13 of 25
    }
    __shared__ int rh[256];
    rh[t] = hits;
    __syncthreads();
    for (int off = 128; off > 0; off >>= 1) {
        if (t < off) rh[t] += rh[t + off];
        __syncthreads();
    }
    if (t == 0) {
        float d = (float)rh[0] * (1.0f / (float)HW_);
        s_out[bc] = sqrtf(a[bc] * d);   // gamma = 0.5
    }
}

// ---------------- Kernel 3: per-batch mask M = U < sigmoid(5(s-0.6*smax)) -----
__global__ void mask_kernel(const float* __restrict__ s, const float* __restrict__ U,
                            int* __restrict__ M) {
    int b = blockIdx.x, c = threadIdx.x;
    __shared__ float rm[256];
    float sc = s[b * C_ + c];
    rm[c] = sc;
    __syncthreads();
    for (int off = 128; off > 0; off >>= 1) {
        if (c < off) rm[c] = fmaxf(rm[c], rm[c + off]);
        __syncthreads();
    }
    float smax = rm[0];
    float prob = 1.0f / (1.0f + expf(-5.0f * (sc - 0.6f * smax)));
    M[b * C_ + c] = (U[b * C_ + c] < prob) ? 1 : 0;
}

// ---------------- Kernel 4a: normalize + convert F -> bf16 (one pass) ---------
__global__ void cvt_kernel(const float* __restrict__ F, const float* __restrict__ invn,
                           unsigned short* __restrict__ Fbf) {
    int gid = blockIdx.x * 256 + threadIdx.x;       // 8 elements per thread
    int bc  = gid >> 9;                              // 512 gids per 4096-elem row
    float scale = invn[bc];
    const float4* src = (const float4*)F;
    float4 v0 = src[gid * 2 + 0];
    float4 v1 = src[gid * 2 + 1];
    unsigned short o[8] = {
        f2bf(v0.x * scale), f2bf(v0.y * scale), f2bf(v0.z * scale), f2bf(v0.w * scale),
        f2bf(v1.x * scale), f2bf(v1.y * scale), f2bf(v1.z * scale), f2bf(v1.w * scale)
    };
    ((uint4*)Fbf)[gid] = *(const uint4*)o;
}

// ------- fragment loaders matching the ISA A/B lane layouts (bf16, K=32) ------
__device__ __forceinline__ v16bf loadA(const unsigned short* row, int k0, int hi) {
    // A 16x32: lanes 0-15 hold K {0..7,16..23}; lanes 16-31 hold K {8..15,24..31}
    U16x16 p;
    p.lo = *(const uint4*)(row + k0 + 8 * hi);
    p.hi = *(const uint4*)(row + k0 + 16 + 8 * hi);
    return __builtin_bit_cast(v16bf, p);
}
__device__ __forceinline__ v16bf loadB(const unsigned short* row, int k0, int hi) {
    // B 32x16: lanes 0-15 hold K {0..7,16..23}; lanes 16-31 hold K {8..15,24..31}
    U16x16 p;
    p.lo = *(const uint4*)(row + k0 + 8 * hi);
    p.hi = *(const uint4*)(row + k0 + 16 + 8 * hi);
    return __builtin_bit_cast(v16bf, p);
}

// ---------------- Kernel 4b: WMMA bf16 GEMM, 2x2 register-blocked -------------
// grid = (36 upper-tri 32x32 block-pairs, 16 batch groups, 2 K-halves), 1 wave.
// part[slice] accumulates sum over this slice's batches/K-range of Fn*Fn^T.
__global__ void __launch_bounds__(32)
gemm_bf16_kernel(const unsigned short* __restrict__ Fbf, float* __restrict__ part) {
    // decode upper-triangular 32x32 block pair (bm <= bn) over an 8x8 grid
    int tt = blockIdx.x, bm = 0;
    while (tt >= 8 - bm) { tt -= 8 - bm; ++bm; }
    int bn = bm + tt;
    int g  = blockIdx.y;
    int kz = blockIdx.z;
    int lane = threadIdx.x;
    int l15 = lane & 15;
    int hi  = lane >> 4;

    int rA0 = bm * 32 + l15, rA1 = rA0 + 16;   // A rows (M) held by this lane
    int rB0 = bn * 32 + l15, rB1 = rB0 + 16;   // B cols (N) held by this lane
    int kbeg = kz * (HW_ / KSPLIT);
    int kend = kbeg + (HW_ / KSPLIT);

    v8f acc00 = {}, acc01 = {}, acc10 = {}, acc11 = {};
    for (int bb = 0; bb < BPG2; ++bb) {
        int b = g * BPG2 + bb;
        const unsigned short* Fa0 = Fbf + ((size_t)(b * C_ + rA0)) * HW_;
        const unsigned short* Fa1 = Fbf + ((size_t)(b * C_ + rA1)) * HW_;
        const unsigned short* Fb0 = Fbf + ((size_t)(b * C_ + rB0)) * HW_;
        const unsigned short* Fb1 = Fbf + ((size_t)(b * C_ + rB1)) * HW_;
        for (int k0 = kbeg; k0 < kend; k0 += 32) {
            v16bf A0 = loadA(Fa0, k0, hi);
            v16bf A1 = loadA(Fa1, k0, hi);
            v16bf Bm0 = loadB(Fb0, k0, hi);
            v16bf Bm1 = loadB(Fb1, k0, hi);
            // 4 independent accumulation chains -> XDL stays pipelined
            acc00 = __builtin_amdgcn_wmma_f32_16x16x32_bf16(false, A0, false, Bm0, (short)0, acc00, false, false);
            acc01 = __builtin_amdgcn_wmma_f32_16x16x32_bf16(false, A0, false, Bm1, (short)0, acc01, false, false);
            acc10 = __builtin_amdgcn_wmma_f32_16x16x32_bf16(false, A1, false, Bm0, (short)0, acc10, false, false);
            acc11 = __builtin_amdgcn_wmma_f32_16x16x32_bf16(false, A1, false, Bm1, (short)0, acc11, false, false);
        }
    }
    // store tiles; skip the lower-triangle duplicate tile on diagonal blocks
    float* out = part + ((size_t)g * KSPLIT + kz) * (C_ * C_);
#pragma unroll
    for (int i = 0; i < 8; ++i) {
        int Mi = i + 8 * hi;                 // C/D: VGPR i -> M=i+8*hi, N=lane&15
        int r0 = bm * 32 + Mi, r1 = r0 + 16;
        int c0 = bn * 32 + l15, c1 = c0 + 16;
        out[r0 * C_ + c0] = acc00[i];
        out[r0 * C_ + c1] = acc01[i];
        if (bm != bn) out[r1 * C_ + c0] = acc10[i];
        out[r1 * C_ + c1] = acc11[i];
    }
}

// ---------------- Kernel 4 (fallback): fused convert + WMMA GEMM, full grid ---
__global__ void __launch_bounds__(32)
gemm_fused_kernel(const float* __restrict__ F, const float* __restrict__ invn,
                  float* __restrict__ part) {
    int tm = blockIdx.x, tn = blockIdx.y, g = blockIdx.z;
    int lane = threadIdx.x;
    int l15 = lane & 15;
    int hi  = lane >> 4;
    int rowA = tm * 16 + l15;
    int rowB = tn * 16 + l15;

    v8f acc = {};
    for (int bb = 0; bb < BPGF; ++bb) {
        int b = g * BPGF + bb;
        const float* Fa = F + ((size_t)(b * C_ + rowA)) * HW_;
        const float* Fb = F + ((size_t)(b * C_ + rowB)) * HW_;
        float sa = invn[b * C_ + rowA];
        float sb = invn[b * C_ + rowB];
        for (int k0 = 0; k0 < HW_; k0 += 32) {
            const float4* a0 = (const float4*)(Fa + k0 + 8 * hi);
            const float4* a1 = (const float4*)(Fa + k0 + 16 + 8 * hi);
            float4 av[4] = { a0[0], a0[1], a1[0], a1[1] };
            const float4* b0 = (const float4*)(Fb + k0 + 8 * hi);
            const float4* b1 = (const float4*)(Fb + k0 + 16 + 8 * hi);
            float4 bv[4] = { b0[0], b0[1], b1[0], b1[1] };
            v16u au, bu;
            const float* ap = (const float*)av;
            const float* bp = (const float*)bv;
#pragma unroll
            for (int i = 0; i < 16; ++i) {
                au[i] = f2bf(ap[i] * sa);
                bu[i] = f2bf(bp[i] * sb);
            }
            v16bf A  = __builtin_bit_cast(v16bf, au);
            v16bf Bm = __builtin_bit_cast(v16bf, bu);
            acc = __builtin_amdgcn_wmma_f32_16x16x32_bf16(
                      false, A, false, Bm, (short)0, acc, false, false);
        }
    }
    float* out = part + (size_t)g * (C_ * C_);
#pragma unroll
    for (int i = 0; i < 8; ++i) {
        int Mi = i + 8 * hi;
        out[(tm * 16 + Mi) * C_ + tn * 16 + l15] = acc[i];
    }
}

// ---------------- Kernel 5: S_new = 0.999*S + 0.001*mean_b(S_batch) -----------
// Reads only the upper triangle of part (exactly symmetric by construction).
__global__ void snew_kernel(const float* __restrict__ S, const float* __restrict__ part,
                            float* __restrict__ S_new, int nslices) {
    int idx = blockIdx.x * 256 + threadIdx.x;
    int r = idx >> 8, c = idx & 255;
    int rr = min(r, c), cc = max(r, c);
    int src = rr * C_ + cc;
    float acc = 0.f;
    for (int sl = 0; sl < nslices; ++sl) acc += part[(size_t)sl * (C_ * C_) + src];
    S_new[idx] = 0.999f * S[idx] + (0.001f / (float)B_) * acc;
}

// ---------------- Kernel 6: build per-batch permutation -----------------------
__global__ void perm_kernel(const float* __restrict__ S_new, const int* __restrict__ M,
                            int* __restrict__ perm) {
    int b = blockIdx.x, c = threadIdx.x;
    __shared__ int sm[256], sj[256], sp[256];
    __shared__ int flags[2];
    int mc = M[b * C_ + c];
    float best = -INFINITY;
    int bestj = 0;
    const float* row = S_new + c * C_;
    for (int j = 0; j < C_; ++j) {
        if (!M[b * C_ + j]) {
            float v = row[j];
            if (v > best) { best = v; bestj = j; }   // first max wins (strict >)
        }
    }
    sm[c] = mc;
    sj[c] = bestj;
    sp[c] = mc ? bestj : c;
    __syncthreads();
    if (c == 0) {
        int anySel = 0, anyUnsel = 0;
        for (int i = 0; i < C_; ++i) { anySel |= sm[i]; anyUnsel |= !sm[i]; }
        for (int i = 0; i < C_; ++i) if (sm[i]) sp[sj[i]] = i;   // last-wins scatter
        flags[0] = anySel; flags[1] = anyUnsel;
    }
    __syncthreads();
    int valid = flags[0] && flags[1];
    perm[b * C_ + c] = valid ? sp[c] : c;
}

// ---------------- Kernel 7: channel-permuted gather ---------------------------
__global__ void gather_kernel(const float* __restrict__ F, const int* __restrict__ perm,
                              float* __restrict__ out) {
    size_t e = (size_t)blockIdx.x * blockDim.x + threadIdx.x;  // float4 index
    int t = (int)(e & 1023);
    int c = (int)((e >> 10) & 255);
    int b = (int)(e >> 18);
    int pc = perm[b * C_ + c];
    const float4* src = (const float4*)F;
    float4* dst = (float4*)out;
    dst[e] = src[((size_t)(b * C_ + pc) << 10) + (size_t)t];
}

extern "C" void kernel_launch(void* const* d_in, const int* in_sizes, int n_in,
                              void* d_out, int out_size, void* d_ws, size_t ws_size,
                              hipStream_t stream) {
    const float* F = (const float*)d_in[0];   // [32,256,64,64]
    const float* S = (const float*)d_in[1];   // [256,256]
    const float* U = (const float*)d_in[2];   // [32,256]
    float* out = (float*)d_out;

    // workspace layout (floats)
    float* ws   = (float*)d_ws;
    float* a    = ws;                          // 8192
    float* invn = ws + 8192;                   // 8192
    float* mx   = ws + 16384;                  // 8192
    float* s    = ws + 24576;                  // 8192
    int*   M    = (int*)(ws + 32768);          // 8192
    int*   perm = (int*)(ws + 40960);          // 8192
    float* part = ws + 49152;                  // up to 32*65536 f = 8 MiB
    float* Snew = ws + 49152 + NSLICE * 65536; // 65536 f
    const size_t SMALL_BYTES = 8912896;        // 8.5 MiB (> end of Snew), Fbf start
    unsigned short* Fbf = (unsigned short*)((char*)d_ws + SMALL_BYTES);
    const size_t NEED_BIG = SMALL_BYTES + (size_t)B_ * C_ * HW_ * sizeof(unsigned short);

    stats_kernel  <<<dim3(B_ * C_), dim3(256), 0, stream>>>(F, a, invn, mx);
    density_kernel<<<dim3(B_ * C_), dim3(256), 0, stream>>>(F, a, mx, s);
    mask_kernel   <<<dim3(B_),      dim3(256), 0, stream>>>(s, U, M);

    int nslices;
    if (ws_size >= NEED_BIG) {
        // fast path: one-pass bf16 normalize, then 2x2-blocked load+WMMA GEMM
        // on upper-triangular 32x32 blocks (result exactly symmetric)
        cvt_kernel      <<<dim3((B_ * C_ * HW_) / (256 * 8)), dim3(256), 0, stream>>>(F, invn, Fbf);
        gemm_bf16_kernel<<<dim3(NPAIR, NGRP, KSPLIT), dim3(32), 0, stream>>>(Fbf, part);
        nslices = NSLICE;
    } else {
        gemm_fused_kernel<<<dim3(16, 16, NPARTF), dim3(32), 0, stream>>>(F, invn, part);
        nslices = NPARTF;
    }

    snew_kernel   <<<dim3(C_), dim3(256), 0, stream>>>(S, part, Snew, nslices);
    perm_kernel   <<<dim3(B_), dim3(256), 0, stream>>>(Snew, M, perm);
    gather_kernel <<<dim3((B_ * C_ * (HW_ / 4)) / 256), dim3(256), 0, stream>>>(F, perm, out);
}
